// GATModel3Pooled_70557722739061
// MI455X (gfx1250) — compile-verified, matrix operand order checked by reading, not run on hardware
//
#include <hip/hip_runtime.h>
#include <hip/hip_bf16.h>
#include <math.h>

// ---------------------------------------------------------------------------
// Problem constants (match reference setup_inputs)
// ---------------------------------------------------------------------------
#define NB   128          // graphs
#define NPG  64           // nodes per graph
#define N0   (NB * NPG)   // 8192 nodes
#define EPG  1024
#define EE   (NB * EPG)   // 131072 edges
#define NES  64
#define EES  32
#define DT   512          // hidden width h*c for every conv
#define K1   32
#define K2   16
#define K3   8
#define BN_EPS 1e-5f
#define LPAD 40           // padded LDS row (bf16 elems): 80B = 20 banks, conflict-free b128

typedef __bf16 v16bf __attribute__((ext_vector_type(16)));
typedef __bf16 v8bf  __attribute__((ext_vector_type(8)));
typedef float  v8f   __attribute__((ext_vector_type(8)));

#define WMMA_BF16 __builtin_amdgcn_wmma_f32_16x16x32_bf16

// Fragment layout (ISA 7.12.2, 16-bit A/B): lane(half,mn) needs two contiguous
// 8-elem runs of its row: k in [half*8, half*8+8) and [16+half*8, ...+8).
__device__ __forceinline__ v16bf load_frag(const __bf16* rowp, int half) {
  v8bf lo = *(const v8bf*)(rowp + half * 8);
  v8bf hi = *(const v8bf*)(rowp + 16 + half * 8);
  return __builtin_shufflevector(lo, hi, 0, 1, 2, 3, 4, 5, 6, 7,
                                 8, 9, 10, 11, 12, 13, 14, 15);
}

// ---------------------------------------------------------------------------
// Tiled bf16 WMMA GEMM:  C[M,N] = A[M,K] @ W[K,N] + bias  (optional relu)
// M % 64 == 0, N % 64 == 0, K % 32 == 0. Block = 256 thr = 8 waves,
// tile 64x64, each wave computes 16x32 (two 16x16x32 WMMAs per k-step).
// A staged row-major, B staged transposed (Bt[n][k]) -> all fragment loads
// are 16B ds_load_b128.
// ---------------------------------------------------------------------------
__global__ __launch_bounds__(256)
void wmma_gemm_bias(const float* __restrict__ A, const float* __restrict__ W,
                    const float* __restrict__ bias, float* __restrict__ C,
                    int M, int K, int N, int relu) {
  __shared__ __align__(16) __bf16 As[64][LPAD];
  __shared__ __align__(16) __bf16 Bt[64][LPAD];   // Bt[n][k]
  const int tid  = threadIdx.x;
  const int lane = tid & 31;
  const int wid  = tid >> 5;
  const int wr   = wid >> 1;   // 0..3 : wave row
  const int wc   = wid & 1;    // 0..1 : wave col
  const int rowBase = blockIdx.x * 64;
  const int colBase = blockIdx.y * 64;
  const int half = lane >> 4;
  const int mn   = lane & 15;

  v8f c0 = {}; v8f c1 = {};
  for (int k0 = 0; k0 < K; k0 += 32) {
    // stage A tile 64x32 : 512 float4 loads (2 per thread)
    for (int i = tid; i < 64 * 8; i += 256) {
      int r = i >> 3, c4 = (i & 7) * 4;
      float4 v = *(const float4*)&A[(size_t)(rowBase + r) * K + k0 + c4];
      __bf16* d = &As[r][c4];
      d[0] = (__bf16)v.x; d[1] = (__bf16)v.y; d[2] = (__bf16)v.z; d[3] = (__bf16)v.w;
    }
    // stage B tile 32x64 transposed : 512 float4 loads (2 per thread)
    for (int i = tid; i < 32 * 16; i += 256) {
      int r = i >> 4, c4 = (i & 15) * 4;
      float4 v = *(const float4*)&W[(size_t)(k0 + r) * N + colBase + c4];
      Bt[c4 + 0][r] = (__bf16)v.x; Bt[c4 + 1][r] = (__bf16)v.y;
      Bt[c4 + 2][r] = (__bf16)v.z; Bt[c4 + 3][r] = (__bf16)v.w;
    }
    // prefetch next K tile while we compute (global_prefetch_b8)
    if (k0 + 32 < K) {
      __builtin_prefetch(&A[(size_t)(rowBase + (tid & 63)) * K + k0 + 32 + (tid >> 6) * 8], 0, 1);
      __builtin_prefetch(&W[(size_t)(k0 + 32 + (tid & 31)) * N + colBase + (tid >> 5) * 8], 0, 1);
    }
    __syncthreads();
    v16bf a  = load_frag(&As[wr * 16 + mn][0], half);
    v16bf b0 = load_frag(&Bt[wc * 32 + mn][0], half);
    v16bf b1 = load_frag(&Bt[wc * 32 + 16 + mn][0], half);
    c0 = WMMA_BF16(false, a, false, b0, (short)0, c0, false, false);
    c1 = WMMA_BF16(false, a, false, b1, (short)0, c1, false, false);
    __syncthreads();
  }
  const int col0 = colBase + wc * 32 + mn;
  const int col1 = col0 + 16;
  const float bb0 = bias ? bias[col0] : 0.0f;
  const float bb1 = bias ? bias[col1] : 0.0f;
#pragma unroll
  for (int r = 0; r < 8; ++r) {
    int row = rowBase + wr * 16 + r + half * 8;
    float v0 = c0[r] + bb0;
    float v1 = c1[r] + bb1;
    if (relu) { v0 = fmaxf(v0, 0.0f); v1 = fmaxf(v1, 0.0f); }
    C[(size_t)row * N + col0] = v0;
    C[(size_t)row * N + col1] = v1;
  }
}

// ---------------------------------------------------------------------------
// GATv2 edge-logit kernel. One wave per 16-edge tile.
//   m   = leaky_relu(xl[src] + xr[dst] + ea@We)   (ea@We via WMMA, K=32)
//   logit[e,h] = sum_c m[e,h,c] * att[h,c]   (masked -> -1e9)
// We (32x512) staged transpose-packed in LDS -> B fragments are ds_load_b128.
// A fragment comes straight from global as 4x float4 per lane.
// ---------------------------------------------------------------------------
__global__ __launch_bounds__(256)
void edge_logit_kernel(const float* __restrict__ ea,   // [E,32]
                       const float* __restrict__ We,   // [32,512]
                       const float* __restrict__ att,  // [heads, 512/heads]
                       const float* __restrict__ xl,   // [n,512]
                       const float* __restrict__ xr,   // [n,512]
                       const int* __restrict__ ns, const int* __restrict__ nd,
                       const int* __restrict__ msk,    // null => all valid
                       float* __restrict__ logit,      // [E, heads]
                       int heads) {
  const int cdim = DT / heads;
  __shared__ __align__(16) __bf16 Wt[DT][LPAD];   // Wt[n][k], 40KB
  __shared__ float atts[DT];
  const int tid = threadIdx.x;
  for (int i = tid; i < 32 * DT; i += 256) {
    int k = i >> 9, n = i & (DT - 1);
    Wt[n][k] = (__bf16)We[i];
  }
  for (int i = tid; i < DT; i += 256) atts[i] = att[i];
  __syncthreads();

  const int lane = tid & 31, wid = tid >> 5;
  const int e0   = (blockIdx.x * 8 + wid) * 16;   // grid sized so e0 < E always
  const int half = lane >> 4, nn = lane & 15;

  int nsr[8], ndr[8], mr[8];
#pragma unroll
  for (int r = 0; r < 8; ++r) {
    int e = e0 + r + 8 * half;
    nsr[r] = ns[e]; ndr[r] = nd[e];
    mr[r] = msk ? msk[e] : 1;
  }
  // A fragment: edge-feature row, two 8-float runs -> four float4 loads
  const float* er = ea + (size_t)(e0 + nn) * EES;
  float4 a0 = *(const float4*)(er + half * 8);
  float4 a1 = *(const float4*)(er + half * 8 + 4);
  float4 a2 = *(const float4*)(er + 16 + half * 8);
  float4 a3 = *(const float4*)(er + 16 + half * 8 + 4);
  v16bf a;
  a[0] = (__bf16)a0.x; a[1] = (__bf16)a0.y; a[2]  = (__bf16)a0.z; a[3]  = (__bf16)a0.w;
  a[4] = (__bf16)a1.x; a[5] = (__bf16)a1.y; a[6]  = (__bf16)a1.z; a[7]  = (__bf16)a1.w;
  a[8] = (__bf16)a2.x; a[9] = (__bf16)a2.y; a[10] = (__bf16)a2.z; a[11] = (__bf16)a2.w;
  a[12] = (__bf16)a3.x; a[13] = (__bf16)a3.y; a[14] = (__bf16)a3.z; a[15] = (__bf16)a3.w;

  float acc[2][8] = {};
#pragma unroll 1
  for (int ch = 0; ch < DT / 16; ++ch) {
    const int n0 = ch * 16;
    v16bf b = load_frag(&Wt[n0 + nn][0], half);
    v8f c = {};
    c = WMMA_BF16(false, a, false, b, (short)0, c, false, false);
    const int   j  = n0 + nn;         // channel
    const int   hd = n0 / cdim;       // head (chunk lies within one head)
    const float aj = atts[j];
#pragma unroll
    for (int r = 0; r < 8; ++r) {
      float v2 = c[r] + xl[(size_t)nsr[r] * DT + j] + xr[(size_t)ndr[r] * DT + j];
      v2 = (v2 > 0.0f) ? v2 : 0.2f * v2;   // leaky_relu 0.2
      acc[hd][r] += v2 * aj;
    }
  }
  // reduce across the 16 lanes of each half
#pragma unroll
  for (int hd = 0; hd < 2; ++hd)
#pragma unroll
    for (int r = 0; r < 8; ++r) {
      float v = acc[hd][r];
      v += __shfl_xor(v, 1, 32);
      v += __shfl_xor(v, 2, 32);
      v += __shfl_xor(v, 4, 32);
      v += __shfl_xor(v, 8, 32);
      acc[hd][r] = v;
    }
  if (nn == 0) {
#pragma unroll
    for (int r = 0; r < 8; ++r) {
      int e = e0 + r + 8 * half;
      for (int hd = 0; hd < heads; ++hd)
        logit[(size_t)e * heads + hd] = mr[r] ? acc[hd][r] : -1e9f;
    }
  }
}

// ---------------------------------------------------------------------------
// Segment softmax helpers (atomics)
// ---------------------------------------------------------------------------
__device__ __forceinline__ void atomicMaxF(float* addr, float val) {
  if (val >= 0.0f) atomicMax((int*)addr, __float_as_int(val));
  else             atomicMin((unsigned int*)addr, (unsigned int)__float_as_int(val));
}

__global__ void fill_f32_kernel(float* p, float v, int n) {
  int i = blockIdx.x * blockDim.x + threadIdx.x;
  if (i < n) p[i] = v;
}

__global__ void seg_max_kernel(const float* __restrict__ logit, const int* __restrict__ nd,
                               float* __restrict__ mx, int E, int heads) {
  int i = blockIdx.x * blockDim.x + threadIdx.x;
  if (i >= E * heads) return;
  int e = i / heads, hd = i - e * heads;
  atomicMaxF(&mx[nd[e] * heads + hd], logit[i]);
}

__global__ void mx_fix_kernel(float* mx, int n) {
  int i = blockIdx.x * blockDim.x + threadIdx.x;
  if (i < n && mx[i] <= -1e29f) mx[i] = 0.0f;   // nodes with no in-edges
}

__global__ void ex_den_kernel(const float* __restrict__ logit, const int* __restrict__ nd,
                              const int* __restrict__ msk, const float* __restrict__ mx,
                              float* __restrict__ ex, float* __restrict__ den, int E, int heads) {
  int i = blockIdx.x * blockDim.x + threadIdx.x;
  if (i >= E * heads) return;
  int e = i / heads, hd = i - e * heads;
  float m = msk ? (msk[e] ? 1.0f : 0.0f) : 1.0f;
  float v = __expf(logit[i] - mx[nd[e] * heads + hd]) * m;
  ex[i] = v;
  atomicAdd(&den[nd[e] * heads + hd], v);
}

__global__ void out_init_kernel(float4* __restrict__ out, const float4* __restrict__ bias,
                                int total4) {
  int i = blockIdx.x * blockDim.x + threadIdx.x;
  if (i < total4) out[i] = bias[i & (DT / 4 - 1)];
}

// 4 channels per thread: float4 gather of xl, 4 atomic f32 adds into out.
__global__ void scatter_kernel(const float* __restrict__ ex, const float* __restrict__ den,
                               const int* __restrict__ ns, const int* __restrict__ nd,
                               const float* __restrict__ xl, float* __restrict__ out,
                               int heads) {
  size_t i = (size_t)blockIdx.x * blockDim.x + threadIdx.x;
  if (i >= (size_t)EE * (DT / 4)) return;
  int e  = (int)(i >> 7);                 // / (DT/4)
  int j4 = ((int)i & (DT / 4 - 1)) * 4;   // first channel of 4
  int hd = j4 / (DT / heads);             // 4-chunk lies within one head
  float d = den[nd[e] * heads + hd];
  float w = ex[(size_t)e * heads + hd] / (d > 0.0f ? d : 1.0f);
  if (w == 0.0f) return;
  float4 v = *(const float4*)&xl[(size_t)ns[e] * DT + j4];
  float* o = &out[(size_t)nd[e] * DT + j4];
  atomicAdd(o + 0, w * v.x);
  atomicAdd(o + 1, w * v.y);
  atomicAdd(o + 2, w * v.z);
  atomicAdd(o + 3, w * v.w);
}

__global__ void bn_relu_kernel(float* __restrict__ x, const float* __restrict__ g,
                               const float* __restrict__ b, const float* __restrict__ rm,
                               const float* __restrict__ rv, int total4) {
  int i = blockIdx.x * blockDim.x + threadIdx.x;
  if (i >= total4) return;
  int j = (i & (DT / 4 - 1)) * 4;
  float4 v = *(float4*)&x[(size_t)i * 4];
  float4 r;
  r.x = fmaxf((v.x - rm[j + 0]) * (g[j + 0] * rsqrtf(rv[j + 0] + BN_EPS)) + b[j + 0], 0.0f);
  r.y = fmaxf((v.y - rm[j + 1]) * (g[j + 1] * rsqrtf(rv[j + 1] + BN_EPS)) + b[j + 1], 0.0f);
  r.z = fmaxf((v.z - rm[j + 2]) * (g[j + 2] * rsqrtf(rv[j + 2] + BN_EPS)) + b[j + 2], 0.0f);
  r.w = fmaxf((v.w - rm[j + 3]) * (g[j + 3] * rsqrtf(rv[j + 3] + BN_EPS)) + b[j + 3], 0.0f);
  *(float4*)&x[(size_t)i * 4] = r;
}

// ---------------------------------------------------------------------------
// TopK pooling
// ---------------------------------------------------------------------------
__global__ void invnorm_kernel(const float* __restrict__ w, float* out, int n) {
  __shared__ float s[256];
  float a = 0.0f;
  for (int i = threadIdx.x; i < n; i += 256) a += w[i] * w[i];
  s[threadIdx.x] = a;
  __syncthreads();
  for (int off = 128; off > 0; off >>= 1) {
    if (threadIdx.x < off) s[threadIdx.x] += s[threadIdx.x + off];
    __syncthreads();
  }
  if (threadIdx.x == 0) out[0] = rsqrtf(s[0]);
}

__global__ void score_kernel(const float* __restrict__ x, const float* __restrict__ w,
                             const float* __restrict__ invn, float* __restrict__ score, int n) {
  int gw = (blockIdx.x * blockDim.x + threadIdx.x) >> 5;
  int lane = threadIdx.x & 31;
  if (gw >= n) return;
  float a = 0.0f;
  for (int j = lane; j < DT; j += 32) a += x[(size_t)gw * DT + j] * w[j];
  for (int off = 16; off > 0; off >>= 1) a += __shfl_xor(a, off, 32);
  if (lane == 0) score[gw] = a * invn[0];
}

__global__ __launch_bounds__(64)
void topk_pool_kernel(const float* __restrict__ x, const float* __restrict__ score,
                      float* __restrict__ xn, int* __restrict__ mapping,
                      int nper, int k) {
  __shared__ float sc[64];
  __shared__ int   selOld[64];
  __shared__ float selScale[64];
  const int g = blockIdx.x, tid = threadIdx.x;
  if (tid < nper) sc[tid] = score[g * nper + tid];
  __syncthreads();
  if (tid < nper) {
    float si = sc[tid];
    int r = 0;
    for (int j = 0; j < nper; ++j) {
      float sj = sc[j];
      if (sj > si || (sj == si && j < tid)) ++r;   // stable top_k ordering
    }
    if (r < k) {
      mapping[g * nper + tid] = g * k + r;
      selOld[r]   = tid;
      selScale[r] = tanhf(si);
    } else {
      mapping[g * nper + tid] = -1;
    }
  }
  __syncthreads();
  for (int t = tid; t < k * (DT / 4); t += 64) {
    int rr = t / (DT / 4), j4 = (t - rr * (DT / 4)) * 4;
    float4 v = *(const float4*)&x[((size_t)g * nper + selOld[rr]) * DT + j4];
    float s = selScale[rr];
    float4 o = { v.x * s, v.y * s, v.z * s, v.w * s };
    *(float4*)&xn[((size_t)g * k + rr) * DT + j4] = o;
  }
}

__global__ void edge_remap_kernel(const int* __restrict__ ns_o, const int* __restrict__ nd_o,
                                  const int* __restrict__ msk_o, const int* __restrict__ mapping,
                                  int* __restrict__ ns_n, int* __restrict__ nd_n,
                                  int* __restrict__ msk_n, int E) {
  int i = blockIdx.x * blockDim.x + threadIdx.x;
  if (i >= E) return;
  int a = mapping[ns_o[i]];
  int b = mapping[nd_o[i]];
  int valid = (msk_o ? msk_o[i] : 1) && (a >= 0) && (b >= 0);
  ns_n[i]  = valid ? a : 0;
  nd_n[i]  = valid ? b : 0;
  msk_n[i] = valid;
}

// ---------------------------------------------------------------------------
// Readout
// ---------------------------------------------------------------------------
__global__ void mean_pool_kernel(const float* __restrict__ x, float* __restrict__ g, int kk) {
  int i = blockIdx.x * blockDim.x + threadIdx.x;
  if (i >= NB * DT) return;
  int b = i >> 9, j = i & (DT - 1);
  float s = 0.0f;
  for (int r = 0; r < kk; ++r) s += x[((size_t)b * kk + r) * DT + j];
  g[i] = s / (float)kk;
}

__global__ void head_kernel(const float* __restrict__ x, const float* __restrict__ w,
                            const float* __restrict__ b, float* __restrict__ out, int K) {
  int gw = (blockIdx.x * blockDim.x + threadIdx.x) >> 5;
  int lane = threadIdx.x & 31;
  if (gw >= NB) return;
  float a = 0.0f;
  for (int j = lane; j < K; j += 32) a += x[(size_t)gw * K + j] * w[j];
  for (int off = 16; off > 0; off >>= 1) a += __shfl_xor(a, off, 32);
  if (lane == 0) out[gw] = 1.0f / (1.0f + __expf(-(a + b[0])));
}

// ---------------------------------------------------------------------------
// Host orchestration
// ---------------------------------------------------------------------------
struct ConvP { const float *Wl, *bl, *Wr, *br, *We, *att, *bias; };

static inline int cdiv(long long a, long long b) { return (int)((a + b - 1) / b); }

static void run_conv(const float* xin, int n, int Din, int heads,
                     const ConvP& p, const float* ea,
                     const int* ns, const int* nd, const int* msk,
                     float* f_xl, float* f_xr, float* f_logit, float* f_ex,
                     float* f_mx, float* f_den, float* f_out, hipStream_t stream) {
  dim3 gg(n / 64, DT / 64);
  wmma_gemm_bias<<<gg, 256, 0, stream>>>(xin, p.Wl, p.bl, f_xl, n, Din, DT, 0);
  wmma_gemm_bias<<<gg, 256, 0, stream>>>(xin, p.Wr, p.br, f_xr, n, Din, DT, 0);
  edge_logit_kernel<<<EE / 16 / 8, 256, 0, stream>>>(ea, p.We, p.att, f_xl, f_xr,
                                                     ns, nd, msk, f_logit, heads);
  int nh = n * heads;
  fill_f32_kernel<<<cdiv(nh, 256), 256, 0, stream>>>(f_mx, -1e30f, nh);
  fill_f32_kernel<<<cdiv(nh, 256), 256, 0, stream>>>(f_den, 0.0f, nh);
  seg_max_kernel<<<cdiv((long long)EE * heads, 256), 256, 0, stream>>>(f_logit, nd, f_mx, EE, heads);
  mx_fix_kernel<<<cdiv(nh, 256), 256, 0, stream>>>(f_mx, nh);
  ex_den_kernel<<<cdiv((long long)EE * heads, 256), 256, 0, stream>>>(f_logit, nd, msk, f_mx,
                                                                      f_ex, f_den, EE, heads);
  out_init_kernel<<<cdiv((long long)n * DT / 4, 256), 256, 0, stream>>>(
      (float4*)f_out, (const float4*)p.bias, n * DT / 4);
  scatter_kernel<<<cdiv((long long)EE * (DT / 4), 256), 256, 0, stream>>>(f_ex, f_den, ns, nd,
                                                                          f_xl, f_out, heads);
}

extern "C" void kernel_launch(void* const* d_in, const int* in_sizes, int n_in,
                              void* d_out, int out_size, void* d_ws, size_t ws_size,
                              hipStream_t stream) {
  (void)in_sizes; (void)n_in; (void)out_size; (void)ws_size;

  const float* x   = (const float*)d_in[0];
  const int*   eix = (const int*)d_in[1];
  const float* ea  = (const float*)d_in[2];
  const int* src = eix;
  const int* dst = eix + EE;

  ConvP cp[3];
  for (int i = 0; i < 3; ++i) {
    int b0 = 4 + i * 7;
    cp[i] = { (const float*)d_in[b0],     (const float*)d_in[b0 + 1],
              (const float*)d_in[b0 + 2], (const float*)d_in[b0 + 3],
              (const float*)d_in[b0 + 4], (const float*)d_in[b0 + 5],
              (const float*)d_in[b0 + 6] };
  }
  const float* bnp[3][4];
  for (int i = 0; i < 3; ++i)
    for (int j = 0; j < 4; ++j) bnp[i][j] = (const float*)d_in[25 + i * 4 + j];
  const float* pw[3] = { (const float*)d_in[37], (const float*)d_in[38], (const float*)d_in[39] };
  const float* l1W = (const float*)d_in[40]; const float* l1b = (const float*)d_in[41];
  const float* l2W = (const float*)d_in[42]; const float* l2b = (const float*)d_in[43];
  const float* l3W = (const float*)d_in[44]; const float* l3b = (const float*)d_in[45];
  float* out = (float*)d_out;

  // ---- workspace carve-up ----
  char* wsb = (char*)d_ws;
  size_t off = 0;
  auto take = [&](size_t elems) -> void* {
    void* p = wsb + off;
    off += (elems * 4 + 255) & ~(size_t)255;
    return p;
  };
  float* f_xl    = (float*)take((size_t)N0 * DT);
  float* f_xr    = (float*)take((size_t)N0 * DT);
  float* f_h     = (float*)take((size_t)N0 * DT);   // conv output / bn
  float* f_x2    = (float*)take((size_t)N0 * DT);   // pooled features
  float* f_logit = (float*)take((size_t)EE * 2);
  float* f_ex    = (float*)take((size_t)EE * 2);
  float* f_mx    = (float*)take((size_t)N0 * 2);
  float* f_den   = (float*)take((size_t)N0 * 2);
  float* f_score = (float*)take(N0);
  float* f_scal  = (float*)take(8);
  int*   i_map   = (int*)take(N0);
  int*   i_ns2   = (int*)take(EE);
  int*   i_nd2   = (int*)take(EE);
  int*   i_mk2   = (int*)take(EE);
  int*   i_ns3   = (int*)take(EE);
  int*   i_nd3   = (int*)take(EE);
  int*   i_mk3   = (int*)take(EE);
  float* f_g     = (float*)take((size_t)NB * DT);
  float* f_l1    = (float*)take((size_t)NB * 256);
  float* f_l2    = (float*)take((size_t)NB * 256);

  // ======== level 1 : conv1 (64 -> 512, H=2) + bn1 + relu + pool(64->32) ====
  run_conv(x, N0, NES, 2, cp[0], ea, src, dst, nullptr,
           f_xl, f_xr, f_logit, f_ex, f_mx, f_den, f_h, stream);
  bn_relu_kernel<<<cdiv((long long)N0 * DT / 4, 256), 256, 0, stream>>>(
      f_h, bnp[0][0], bnp[0][1], bnp[0][2], bnp[0][3], N0 * DT / 4);
  invnorm_kernel<<<1, 256, 0, stream>>>(pw[0], f_scal, DT);
  score_kernel<<<N0 / 8, 256, 0, stream>>>(f_h, pw[0], f_scal, f_score, N0);
  topk_pool_kernel<<<NB, 64, 0, stream>>>(f_h, f_score, f_x2, i_map, NPG, K1);
  edge_remap_kernel<<<cdiv(EE, 256), 256, 0, stream>>>(src, dst, nullptr, i_map,
                                                       i_ns2, i_nd2, i_mk2, EE);

  // ======== level 2 : conv2 (512 -> 512, H=2) + bn2 + relu + pool(32->16) ===
  const int n2 = NB * K1;  // 4096
  run_conv(f_x2, n2, DT, 2, cp[1], ea, i_ns2, i_nd2, i_mk2,
           f_xl, f_xr, f_logit, f_ex, f_mx, f_den, f_h, stream);
  bn_relu_kernel<<<cdiv((long long)n2 * DT / 4, 256), 256, 0, stream>>>(
      f_h, bnp[1][0], bnp[1][1], bnp[1][2], bnp[1][3], n2 * DT / 4);
  invnorm_kernel<<<1, 256, 0, stream>>>(pw[1], f_scal, DT);
  score_kernel<<<n2 / 8, 256, 0, stream>>>(f_h, pw[1], f_scal, f_score, n2);
  topk_pool_kernel<<<NB, 64, 0, stream>>>(f_h, f_score, f_x2, i_map, K1, K2);
  edge_remap_kernel<<<cdiv(EE, 256), 256, 0, stream>>>(i_ns2, i_nd2, i_mk2, i_map,
                                                       i_ns3, i_nd3, i_mk3, EE);

  // ======== level 3 : conv3 (512 -> 512, H=1) + bn3 + relu + pool(16->8) ====
  const int n3 = NB * K2;  // 2048
  run_conv(f_x2, n3, DT, 1, cp[2], ea, i_ns3, i_nd3, i_mk3,
           f_xl, f_xr, f_logit, f_ex, f_mx, f_den, f_h, stream);
  bn_relu_kernel<<<cdiv((long long)n3 * DT / 4, 256), 256, 0, stream>>>(
      f_h, bnp[2][0], bnp[2][1], bnp[2][2], bnp[2][3], n3 * DT / 4);
  invnorm_kernel<<<1, 256, 0, stream>>>(pw[2], f_scal, DT);
  score_kernel<<<n3 / 8, 256, 0, stream>>>(f_h, pw[2], f_scal, f_score, n3);
  topk_pool_kernel<<<NB, 64, 0, stream>>>(f_h, f_score, f_x2, i_map, K2, K3);
  // (post-pool3 edges unused by the reference)

  // ======== readout : mean pool + MLP + sigmoid =============================
  mean_pool_kernel<<<cdiv(NB * DT, 256), 256, 0, stream>>>(f_x2, f_g, K3);
  wmma_gemm_bias<<<dim3(NB / 64, 256 / 64), 256, 0, stream>>>(f_g, l1W, l1b, f_l1,
                                                              NB, DT, 256, 1);
  wmma_gemm_bias<<<dim3(NB / 64, 256 / 64), 256, 0, stream>>>(f_l1, l2W, l2b, f_l2,
                                                              NB, 256, 256, 1);
  head_kernel<<<cdiv(NB * 32, 256), 256, 0, stream>>>(f_l2, l3W, l3b, out, 256);
}
